// GAT_LP_14912126451851
// MI455X (gfx1250) — matrix operand order repaired and checked by reading, not compiled, not run on hardware
//
#include <hip/hip_runtime.h>

#define CDIV(a, b) (((a) + (b) - 1) / (b))
#define NEG_SLOPE 0.2f

typedef __attribute__((ext_vector_type(16))) _Float16 v16h;
typedef __attribute__((ext_vector_type(8)))  _Float16 v8h;
typedef __attribute__((ext_vector_type(8)))  float    v8f;

// ---------------------------------------------------------------------------
// helpers
// ---------------------------------------------------------------------------
__device__ __forceinline__ void atomicMaxF(float* addr, float val) {
  // classic sign-split trick; valid when slot initialized to -FLT_MAX-like
  if (val >= 0.0f) atomicMax((int*)addr, __float_as_int(val));
  else             atomicMin((unsigned int*)addr, __float_as_uint(val));
}

__global__ void k_fill_f32(float* __restrict__ p, float v, long long n) {
  long long i = (long long)blockIdx.x * blockDim.x + threadIdx.x;
  long long st = (long long)gridDim.x * blockDim.x;
  for (; i < n; i += st) p[i] = v;
}

// f32 [K,256] row-major  ->  f16 [256,K] (transposed, K-contiguous per column)
__global__ void k_f32_to_f16_T(const float* __restrict__ src, _Float16* __restrict__ dst, int K) {
  int i = blockIdx.x * blockDim.x + threadIdx.x;  // over K*256
  if (i < K * 256) {
    int k = i >> 8, n = i & 255;
    dst[n * K + k] = (_Float16)src[i];
  }
}

// h = emb[x] gathered + cast to f16, [N,128]
__global__ void k_gather_emb_f16(const int* __restrict__ x, const float* __restrict__ emb,
                                 _Float16* __restrict__ out, int N) {
  int i = blockIdx.x * blockDim.x + threadIdx.x;
  if (i < N * 128) {
    int n = i >> 7, d = i & 127;
    out[i] = (_Float16)emb[(long long)x[n] * 128 + d];
  }
}

// ---------------------------------------------------------------------------
// WMMA GEMM:  C[N,256] = A[N,K] (f16, row-major) @ B[K,256] (f16, given
// TRANSPOSED as Bt[256][K]) with f32 accumulate.
// Block: 256 threads (8 waves). Block owns one 16-row M-tile; each wave does
// two 16-col N-tiles. K in {128, 32}, always a multiple of 32.
// ---------------------------------------------------------------------------
template <int K>
__global__ void k_gemm_wmma(const _Float16* __restrict__ A,
                            const _Float16* __restrict__ Bt,
                            float* __restrict__ C, int nrows) {
  __shared__ _Float16 Atile[16][K];
  const int row0 = blockIdx.x * 16;
  const int tid = threadIdx.x;

  // cooperative stage of the 16xK A tile into LDS, 128-bit chunks
  constexpr int CHUNKS = 16 * K / 8;  // 8 halves per chunk
  for (int idx = tid; idx < CHUNKS; idx += 256) {
    int r = (idx * 8) / K;
    int k = (idx * 8) % K;
    int gr = row0 + r;
    v8h val = {};
    if (gr < nrows) val = *(const v8h*)(A + (long long)gr * K + k);
    *(v8h*)&Atile[r][k] = val;
  }
  __syncthreads();

  const int wave = tid >> 5;
  const int lane = tid & 31;
  const int l16 = lane & 15;
  const int hi = lane >> 4;  // 0: lanes 0-15, 1: lanes 16-31
  const bool full = (row0 + 16) <= nrows;

#pragma unroll
  for (int t = 0; t < 2; ++t) {
    const int n0 = (wave * 2 + t) << 4;  // output column tile base
    const _Float16* bcol = Bt + (long long)(n0 + l16) * K;  // this lane's B column
    v8f acc = {};
#pragma unroll
    for (int k0 = 0; k0 < K; k0 += 32) {
      // A fragment (16x32 f16, contiguous k order, same as B): row = l16;
      //  lanes 0-15 : halves 0-15 = K 0-15
      //  lanes 16-31: halves 0-15 = K 16-31
      const int ka = k0 + hi * 16;
      v8h a0 = *(const v8h*)&Atile[l16][ka];        // 16B-aligned ds_load_b128
      v8h a1 = *(const v8h*)&Atile[l16][ka + 8];
      v16h a = __builtin_shufflevector(a0, a1, 0, 1, 2, 3, 4, 5, 6, 7,
                                       8, 9, 10, 11, 12, 13, 14, 15);
      // B fragment (32x16 f16): col = n0+l16;
      //  lanes 0-15 hold K k0+0..15, lanes 16-31 hold K k0+16..31
      //  -> one contiguous 32-byte load from the transposed weight column
      v16h b = *(const v16h*)(bcol + k0 + hi * 16);

      acc = __builtin_amdgcn_wmma_f32_16x16x32_f16(
          /*neg_a=*/false, a, /*neg_b=*/false, b,
          /*c_mod=*/(short)0, acc, /*reuse_a=*/false, /*reuse_b=*/false);
    }
    // D layout: VGPR r -> M = r + hi*8, N = l16
    float* crow = C + (long long)(row0 + hi * 8) * 256 + n0 + l16;
    if (full) {
#pragma unroll
      for (int r = 0; r < 8; ++r) crow[(long long)r * 256] = acc[r];
    } else {
#pragma unroll
      for (int r = 0; r < 8; ++r)
        if (row0 + hi * 8 + r < nrows) crow[(long long)r * 256] = acc[r];
    }
  }
}

// ---------------------------------------------------------------------------
// attention scalars: a_src[n,h] = <xl[n,h,:], att_src[h,:]>, same for dst
// ---------------------------------------------------------------------------
__global__ void k_attn_scores(const float* __restrict__ xl,
                              const float* __restrict__ att_s,
                              const float* __restrict__ att_d,
                              float* __restrict__ a_src, float* __restrict__ a_dst, int N) {
  int i = blockIdx.x * blockDim.x + threadIdx.x;  // over N*8
  if (i < N * 8) {
    int h = i & 7;
    const float* v = xl + (long long)(i >> 3) * 256 + h * 32;
    const float* as = att_s + h * 32;
    const float* ad = att_d + h * 32;
    float ss = 0.f, sd = 0.f;
#pragma unroll
    for (int c = 0; c < 32; ++c) { float xv = v[c]; ss += xv * as[c]; sd += xv * ad[c]; }
    a_src[i] = ss;
    a_dst[i] = sd;
  }
}

// ---------------------------------------------------------------------------
// edge pass 1: e = leaky_relu(a_src[s]+a_dst[d]); store e; atomic segment max
// edge id e < E: graph edge; e >= E: self loop (s = d = e - E)
// ---------------------------------------------------------------------------
__global__ void k_edge_max(const int* __restrict__ ei, int E, int N,
                           const float* __restrict__ a_src, const float* __restrict__ a_dst,
                           float* __restrict__ ebuf, float* __restrict__ m) {
  long long i = (long long)blockIdx.x * blockDim.x + threadIdx.x;  // over (E+N)*8
  if (i < (long long)(E + N) * 8) {
    int h = (int)(i & 7);
    long long e = i >> 3;
    int s, d;
    if (e < E) { s = ei[e]; d = ei[E + e]; } else { s = d = (int)(e - E); }
    float v = a_src[(long long)s * 8 + h] + a_dst[(long long)d * 8 + h];
    v = v > 0.f ? v : NEG_SLOPE * v;
    ebuf[i] = v;
    atomicMaxF(&m[(long long)d * 8 + h], v);
  }
}

// edge pass 2: ee = exp(e - m[d]); overwrite ebuf; atomic segment sum
__global__ void k_edge_expsum(const int* __restrict__ ei, int E, int N,
                              float* __restrict__ ebuf, const float* __restrict__ m,
                              float* __restrict__ ssum) {
  long long i = (long long)blockIdx.x * blockDim.x + threadIdx.x;
  if (i < (long long)(E + N) * 8) {
    int h = (int)(i & 7);
    long long e = i >> 3;
    int d = (e < E) ? ei[E + e] : (int)(e - E);
    float ee = __expf(ebuf[i] - m[(long long)d * 8 + h]);
    ebuf[i] = ee;
    atomicAdd(&ssum[(long long)d * 8 + h], ee);
  }
}

// edge pass 3: agg[d, h, c] += (ee/ssum[d,h]) * xl[s, h, c]
// one thread per (edge, h*32+c)
__global__ void k_edge_agg(const int* __restrict__ ei, int E, int N,
                           const float* __restrict__ ebuf, const float* __restrict__ ssum,
                           const float* __restrict__ xl, float* __restrict__ agg) {
  long long t = (long long)blockIdx.x * blockDim.x + threadIdx.x;  // over (E+N)*256
  if (t < (long long)(E + N) * 256) {
    long long e = t >> 8;
    int j = (int)(t & 255);
    int h = j >> 5;
    int s, d;
    if (e < E) { s = ei[e]; d = ei[E + e]; } else { s = d = (int)(e - E); }
    float alpha = ebuf[e * 8 + h] / ssum[(long long)d * 8 + h];
    atomicAdd(&agg[(long long)d * 256 + j], alpha * xl[(long long)s * 256 + j]);
  }
}

// head mean + bias (+relu) -> f16 activation for next GEMM
__global__ void k_head_mean_relu_f16(const float* __restrict__ agg, const float* __restrict__ bias,
                                     _Float16* __restrict__ out, int N) {
  int i = blockIdx.x * blockDim.x + threadIdx.x;  // over N*32
  if (i < N * 32) {
    int c = i & 31;
    long long n = i >> 5;
    float sum = 0.f;
#pragma unroll
    for (int h = 0; h < 8; ++h) sum += agg[n * 256 + h * 32 + c];
    float v = sum * 0.125f + bias[c];
    out[i] = (_Float16)(v > 0.f ? v : 0.f);
  }
}

// head mean + bias -> f32 final node embedding z
__global__ void k_head_mean_bias(const float* __restrict__ agg, const float* __restrict__ bias,
                                 float* __restrict__ z, int N) {
  int i = blockIdx.x * blockDim.x + threadIdx.x;  // over N*32
  if (i < N * 32) {
    int c = i & 31;
    long long n = i >> 5;
    float sum = 0.f;
#pragma unroll
    for (int h = 0; h < 8; ++h) sum += agg[n * 256 + h * 32 + c];
    z[i] = sum * 0.125f + bias[c];
  }
}

// decode: out[e] = <z[eli0[e]], z[eli1[e]]>
__global__ void k_decode(const int* __restrict__ eli, int EL, const float* __restrict__ z,
                         float* __restrict__ out) {
  int i = blockIdx.x * blockDim.x + threadIdx.x;
  if (i < EL) {
    const float* za = z + (long long)eli[i] * 32;
    const float* zb = z + (long long)eli[EL + i] * 32;
    float s = 0.f;
#pragma unroll
    for (int c = 0; c < 32; ++c) s += za[c] * zb[c];
    out[i] = s;
  }
}

// ---------------------------------------------------------------------------
// launch
// ---------------------------------------------------------------------------
extern "C" void kernel_launch(void* const* d_in, const int* in_sizes, int n_in,
                              void* d_out, int out_size, void* d_ws, size_t ws_size,
                              hipStream_t stream) {
  const int*   x    = (const int*)d_in[0];
  const int*   ei   = (const int*)d_in[1];
  const int*   eli  = (const int*)d_in[2];
  const float* emb  = (const float*)d_in[3];
  const float* W1   = (const float*)d_in[4];
  const float* as1  = (const float*)d_in[5];
  const float* ad1  = (const float*)d_in[6];
  const float* b1   = (const float*)d_in[7];
  const float* W2   = (const float*)d_in[8];
  const float* as2  = (const float*)d_in[9];
  const float* ad2  = (const float*)d_in[10];
  const float* b2   = (const float*)d_in[11];
  float* out = (float*)d_out;

  const int N  = in_sizes[0];
  const int E  = in_sizes[1] / 2;
  const int EL = in_sizes[2] / 2;
  const long long EN8   = (long long)(E + N) * 8;
  const long long EN256 = (long long)(E + N) * 256;

  // workspace carve-out
  char* p = (char*)d_ws;
  auto carve = [&](size_t bytes) -> void* {
    void* r = (void*)p;
    p += (bytes + 255) & ~(size_t)255;
    return r;
  };
  _Float16* hf16 = (_Float16*)carve((size_t)N * 128 * 2);
  _Float16* W1t  = (_Float16*)carve((size_t)128 * 256 * 2);  // [256][128] f16
  _Float16* W2t  = (_Float16*)carve((size_t)32 * 256 * 2);   // [256][32]  f16
  float* xl   = (float*)carve((size_t)N * 256 * 4);          // reused by both layers
  float* asr  = (float*)carve((size_t)N * 8 * 4);
  float* ads  = (float*)carve((size_t)N * 8 * 4);
  float* m    = (float*)carve((size_t)N * 8 * 4);
  float* ssum = (float*)carve((size_t)N * 8 * 4);
  float* ebuf = (float*)carve((size_t)(E + N) * 8 * 4);
  float* agg  = (float*)carve((size_t)N * 256 * 4);          // reused by both layers
  _Float16* h2 = (_Float16*)carve((size_t)N * 32 * 2);
  float* z    = (float*)carve((size_t)N * 32 * 4);

  const int B256 = 256;

  // weight transpose+conversion, embedding gather
  k_f32_to_f16_T<<<CDIV(128 * 256, B256), B256, 0, stream>>>(W1, W1t, 128);
  k_f32_to_f16_T<<<CDIV(32 * 256, B256), B256, 0, stream>>>(W2, W2t, 32);
  k_gather_emb_f16<<<CDIV(N * 128, B256), B256, 0, stream>>>(x, emb, hf16, N);

  // ---------------- layer 1 ----------------
  k_gemm_wmma<128><<<CDIV(N, 16), 256, 0, stream>>>(hf16, W1t, xl, N);
  k_attn_scores<<<CDIV(N * 8, B256), B256, 0, stream>>>(xl, as1, ad1, asr, ads, N);
  k_fill_f32<<<CDIV((long long)N * 8, B256), B256, 0, stream>>>(m, -3.0e38f, (long long)N * 8);
  k_fill_f32<<<CDIV((long long)N * 8, B256), B256, 0, stream>>>(ssum, 0.f, (long long)N * 8);
  k_fill_f32<<<CDIV((long long)N * 256, B256), B256, 0, stream>>>(agg, 0.f, (long long)N * 256);
  k_edge_max<<<(unsigned)CDIV(EN8, B256), B256, 0, stream>>>(ei, E, N, asr, ads, ebuf, m);
  k_edge_expsum<<<(unsigned)CDIV(EN8, B256), B256, 0, stream>>>(ei, E, N, ebuf, m, ssum);
  k_edge_agg<<<(unsigned)CDIV(EN256, B256), B256, 0, stream>>>(ei, E, N, ebuf, ssum, xl, agg);
  k_head_mean_relu_f16<<<CDIV(N * 32, B256), B256, 0, stream>>>(agg, b1, h2, N);

  // ---------------- layer 2 ----------------
  k_gemm_wmma<32><<<CDIV(N, 16), 256, 0, stream>>>(h2, W2t, xl, N);
  k_attn_scores<<<CDIV(N * 8, B256), B256, 0, stream>>>(xl, as2, ad2, asr, ads, N);
  k_fill_f32<<<CDIV((long long)N * 8, B256), B256, 0, stream>>>(m, -3.0e38f, (long long)N * 8);
  k_fill_f32<<<CDIV((long long)N * 8, B256), B256, 0, stream>>>(ssum, 0.f, (long long)N * 8);
  k_fill_f32<<<CDIV((long long)N * 256, B256), B256, 0, stream>>>(agg, 0.f, (long long)N * 256);
  k_edge_max<<<(unsigned)CDIV(EN8, B256), B256, 0, stream>>>(ei, E, N, asr, ads, ebuf, m);
  k_edge_expsum<<<(unsigned)CDIV(EN8, B256), B256, 0, stream>>>(ei, E, N, ebuf, m, ssum);
  k_edge_agg<<<(unsigned)CDIV(EN256, B256), B256, 0, stream>>>(ei, E, N, ebuf, ssum, xl, agg);
  k_head_mean_bias<<<CDIV(N * 32, B256), B256, 0, stream>>>(agg, b2, z, N);

  // ---------------- decode ----------------
  k_decode<<<CDIV(EL, B256), B256, 0, stream>>>(eli, EL, z, out);
}
